// TransformerBlock_50663434224125
// MI455X (gfx1250) — matrix operand order repaired and checked by reading, not compiled, not run on hardware
//
#include <hip/hip_runtime.h>
#include <hip/hip_bf16.h>

// ---------------------------------------------------------------------------
// Types / helpers
// ---------------------------------------------------------------------------
typedef __attribute__((ext_vector_type(16))) __bf16 v16bf;
typedef __attribute__((ext_vector_type(8)))  float  v8f;

union Frag {
  v16bf v;
  unsigned int u[8];
};

// CDNA5 async global->LDS copy path (ASYNCcnt), if the toolchain declares it.
#if defined(__AMDGCN__) && defined(__has_builtin)
#if __has_builtin(__builtin_amdgcn_global_load_async_to_lds_b128)
#define USE_ASYNC_COPY 1
#endif
#endif
#ifndef USE_ASYNC_COPY
#define USE_ASYNC_COPY 0
#endif

#if USE_ASYNC_COPY
// Builtin signature (from compiler diagnostic): param0 = AS1 pointer to
// int __attribute__((vector_size(16))), param1 = LDS pointer, then 2 ints.
typedef int v4i_t __attribute__((vector_size(16)));
typedef __attribute__((address_space(1))) v4i_t glb_v4i_t;
typedef __attribute__((address_space(3))) v4i_t lds_v4i_t;
#endif

__device__ __forceinline__ unsigned short f2bf(float f) {
  // round-to-nearest-even fp32 -> bf16
  unsigned int u = __float_as_uint(f);
  u += 0x7fffu + ((u >> 16) & 1u);
  return (unsigned short)(u >> 16);
}

__device__ __forceinline__ v8f zero8() {
  v8f z;
#pragma unroll
  for (int i = 0; i < 8; ++i) z[i] = 0.0f;
  return z;
}

__device__ __forceinline__ v8f wmma_bf16(const Frag& a, const Frag& b, v8f c) {
  return __builtin_amdgcn_wmma_f32_16x16x32_bf16(
      /*neg_a=*/false, a.v, /*neg_b=*/false, b.v,
      /*c_mod=*/(short)0, c, /*reuse_a=*/false, /*reuse_b=*/false);
}

// 16-byte global->LDS tile chunk copy (async if available)
__device__ __forceinline__ void copy16(const unsigned short* gsrc,
                                       unsigned short* ldst) {
#if USE_ASYNC_COPY
  __builtin_amdgcn_global_load_async_to_lds_b128(
      (glb_v4i_t*)(size_t)gsrc,
      (lds_v4i_t*)(unsigned int)(size_t)ldst, 0, 0);
#else
  *(uint4*)ldst = *(const uint4*)gsrc;
#endif
}

__device__ __forceinline__ void copy_fence() {
#if USE_ASYNC_COPY
  asm volatile("s_wait_asynccnt 0" ::: "memory");
#endif
}

// ---------------------------------------------------------------------------
// fp32 -> bf16 elementwise convert (weights)
// ---------------------------------------------------------------------------
__global__ void cvt_f32_bf16_kernel(const float* __restrict__ in,
                                    unsigned short* __restrict__ out, int n) {
  int i = blockIdx.x * 256 + threadIdx.x;
  if (i < n) out[i] = f2bf(in[i]);
}

// ---------------------------------------------------------------------------
// LayerNorm over D=512, fp32 in, bf16 out. One block (256 thr) per row.
// ---------------------------------------------------------------------------
__global__ __launch_bounds__(256) void layernorm_bf16_kernel(
    const float* __restrict__ x, const float* __restrict__ g,
    const float* __restrict__ b, unsigned short* __restrict__ out) {
  const int D = 512;
  __shared__ float red[256];
  const int row = blockIdx.x;
  const int t = threadIdx.x;
  const float* xr = x + (size_t)row * D;
  float v0 = xr[t], v1 = xr[t + 256];

  red[t] = v0 + v1;
  __syncthreads();
  for (int st = 128; st > 0; st >>= 1) {
    if (t < st) red[t] += red[t + st];
    __syncthreads();
  }
  float mean = red[0] * (1.0f / D);
  __syncthreads();

  float d0 = v0 - mean, d1 = v1 - mean;
  red[t] = d0 * d0 + d1 * d1;
  __syncthreads();
  for (int st = 128; st > 0; st >>= 1) {
    if (t < st) red[t] += red[t + st];
    __syncthreads();
  }
  float inv = rsqrtf(red[0] * (1.0f / D) + 1e-5f);

  out[(size_t)row * D + t]       = f2bf(d0 * inv * g[t] + b[t]);
  out[(size_t)row * D + t + 256] = f2bf(d1 * inv * g[t + 256] + b[t + 256]);
}

// ---------------------------------------------------------------------------
// Generic bf16 WMMA GEMM:  C[M,N] = A[M,K] @ W[N,K]^T (+bias)(+relu)(+res)
// Block: 256 thr (8 waves), tile 128x128, wave tile 64x32, K step 64.
// LDS: two 128x64 bf16 panels (16 KB each).
// ---------------------------------------------------------------------------
template <bool RELU, bool RES, bool OUTBF>
__global__ __launch_bounds__(256) void gemm_bf16_kernel(
    const unsigned short* __restrict__ A,   // [M,K] bf16
    const unsigned short* __restrict__ Bw,  // [N,K] bf16
    const float* __restrict__ bias,         // [N]
    const float* __restrict__ res,          // [M,N] fp32 (if RES)
    float* __restrict__ outF,               // [M,N] fp32 (if !OUTBF)
    unsigned short* __restrict__ outH,      // [M,N] bf16 (if OUTBF)
    int M, int N, int K) {
  __shared__ __align__(16) unsigned short lA[128 * 64];
  __shared__ __align__(16) unsigned short lB[128 * 64];

  const int tid = threadIdx.x;
  const int wv = tid >> 5;
  const int lane = tid & 31;
  const int m = lane & 15;   // A-row / B-col / C-col within 16-tile
  const int h = lane >> 4;   // lane half

  const int nBase = blockIdx.x * 128;
  const int mBase = blockIdx.y * 128;
  const int waveM = (wv >> 2) * 64;
  const int waveN = (wv & 3) * 32;

  v8f acc[4][2];
#pragma unroll
  for (int mt = 0; mt < 4; ++mt)
#pragma unroll
    for (int nt = 0; nt < 2; ++nt) acc[mt][nt] = zero8();

  for (int k0 = 0; k0 < K; k0 += 64) {
    __syncthreads();
    // stage A and B panels: 128 rows x 64 cols bf16 each (16 KB)
    {
      const unsigned short* srcA = A + (size_t)mBase * K + k0;
      const unsigned short* srcB = Bw + (size_t)nBase * K + k0;
#pragma unroll
      for (int i = 0; i < 4; ++i) {
        int c = tid * 4 + i;       // 1024 chunks of 8 ushorts per panel
        int row = c >> 3;
        int ko = (c & 7) * 8;
        copy16(srcA + (size_t)row * K + ko, &lA[row * 64 + ko]);
        copy16(srcB + (size_t)row * K + ko, &lB[row * 64 + ko]);
      }
    }
    copy_fence();
    __syncthreads();

#pragma unroll
    for (int kc = 0; kc < 2; ++kc) {
      Frag fa[4], fb[2];
#pragma unroll
      for (int mt = 0; mt < 4; ++mt) {
        int base = (waveM + mt * 16 + m) * 64;
#pragma unroll
        for (int v = 0; v < 8; ++v) {
          int k = kc * 32 + ((v & 4) << 2) + h * 8 + (v & 3) * 2;  // A-frag
          fa[mt].u[v] = *(const unsigned int*)&lA[base + k];
        }
      }
#pragma unroll
      for (int nt = 0; nt < 2; ++nt) {
        int base = (waveN + nt * 16 + m) * 64;
#pragma unroll
        for (int v = 0; v < 8; ++v) {
          int k = kc * 32 + ((v & 4) << 2) + h * 8 + (v & 3) * 2;  // B-frag
          fb[nt].u[v] = *(const unsigned int*)&lB[base + k];
        }
      }
#pragma unroll
      for (int mt = 0; mt < 4; ++mt)
#pragma unroll
        for (int nt = 0; nt < 2; ++nt)
          acc[mt][nt] = wmma_bf16(fa[mt], fb[nt], acc[mt][nt]);
    }
  }

  // epilogue
#pragma unroll
  for (int nt = 0; nt < 2; ++nt) {
    int gn = nBase + waveN + nt * 16 + m;
    float bv = bias[gn];
#pragma unroll
    for (int mt = 0; mt < 4; ++mt) {
#pragma unroll
      for (int r = 0; r < 8; ++r) {
        int gm = mBase + waveM + mt * 16 + r + 8 * h;
        float val = acc[mt][nt][r] + bv;
        if (RELU) val = fmaxf(val, 0.0f);
        if (RES) val += res[(size_t)gm * N + gn];
        if (OUTBF) outH[(size_t)gm * N + gn] = f2bf(val);
        else       outF[(size_t)gm * N + gn] = val;
      }
    }
  }
}

// ---------------------------------------------------------------------------
// Flash-style ALiBi attention.
// Grid: (S/64, H, B). Block: 128 thr (4 waves); wave w owns 16 Q rows.
// q/k/v/ctx are bf16 in [B*S, D] layout with column = head*64 + dh.
// ---------------------------------------------------------------------------
__global__ __launch_bounds__(128) void attn_kernel(
    const unsigned short* __restrict__ qg,
    const unsigned short* __restrict__ kg,
    const unsigned short* __restrict__ vg,
    unsigned short* __restrict__ ctxg) {
  const int S = 2048, D = 512, DH = 64;
  __shared__ __align__(16) unsigned short lVT[64 * 64];      // V^T tile (8 KB)
  __shared__ __align__(16) unsigned short lP[4 * 16 * 64];   // per-wave P (8 KB)

  const int tid = threadIdx.x;
  const int wv = tid >> 5;
  const int lane = tid & 31;
  const int m = lane & 15;
  const int h = lane >> 4;

  const int qt = blockIdx.x * 64;
  const int head = blockIdx.y;
  const int b = blockIdx.z;

  const float slope = exp2f(-(float)(head + 1));  // 2^{-8 i / H}, H=8

  // load Q A-fragments once (2 K-chunks over DH=64)
  Frag fq[2];
  {
    const int qrow = qt + wv * 16 + m;
    const unsigned short* qp = qg + ((size_t)(b * S + qrow)) * D + head * DH;
#pragma unroll
    for (int kc = 0; kc < 2; ++kc)
#pragma unroll
      for (int v = 0; v < 8; ++v) {
        int k = kc * 32 + ((v & 4) << 2) + h * 8 + (v & 3) * 2;
        fq[kc].u[v] = *(const unsigned int*)(qp + k);
      }
  }

  v8f ctx[4];
#pragma unroll
  for (int dt = 0; dt < 4; ++dt) ctx[dt] = zero8();
  float Mrow[8], Lrow[8];
#pragma unroll
  for (int r = 0; r < 8; ++r) { Mrow[r] = -3.0e38f; Lrow[r] = 0.0f; }

  unsigned short* pw = &lP[wv * 1024];

  for (int kb0 = 0; kb0 < S; kb0 += 64) {
    __syncthreads();  // previous-iteration lVT readers done
    // stage V^T: V[64 keys][64 dh] -> lVT[dh][key]
#pragma unroll 4
    for (int it = 0; it < 16; ++it) {
      int p = tid + 128 * it;            // 0..2047 dword pairs
      int j = p >> 5;                    // key within block
      int d2 = (p & 31) * 2;             // dh pair
      const unsigned short* vp =
          vg + ((size_t)(b * S + kb0 + j)) * D + head * DH + d2;
      unsigned int val = *(const unsigned int*)vp;
      lVT[d2 * 64 + j]       = (unsigned short)(val & 0xffffu);
      lVT[(d2 + 1) * 64 + j] = (unsigned short)(val >> 16);
    }
    __syncthreads();

    // scores: S16x64 = Q(16x64) @ K(64x64)^T
    v8f sacc[4];
#pragma unroll
    for (int nt = 0; nt < 4; ++nt) {
      v8f a = zero8();
#pragma unroll
      for (int kc = 0; kc < 2; ++kc) {
        Frag fk;
        const int krow = kb0 + nt * 16 + m;  // m acts as B-frag column (key)
        const unsigned short* kp =
            kg + ((size_t)(b * S + krow)) * D + head * DH;
#pragma unroll
        for (int v = 0; v < 8; ++v) {
          int k = kc * 32 + ((v & 4) << 2) + h * 8 + (v & 3) * 2;
          fk.u[v] = *(const unsigned int*)(kp + k);
        }
        a = wmma_bf16(fq[kc], fk, a);
      }
      sacc[nt] = a;
    }

    // online softmax with ALiBi
#pragma unroll
    for (int r = 0; r < 8; ++r) {
      float qpos = (float)(qt + wv * 16 + r + 8 * h);
      float rowmax = -3.0e38f;
#pragma unroll
      for (int nt = 0; nt < 4; ++nt) {
        float kpos = (float)(kb0 + nt * 16 + m);
        float sv = sacc[nt][r] * 0.125f - slope * fabsf(qpos - kpos);
        sacc[nt][r] = sv;
        rowmax = fmaxf(rowmax, sv);
      }
#pragma unroll
      for (int msk = 8; msk >= 1; msk >>= 1)
        rowmax = fmaxf(rowmax, __shfl_xor(rowmax, msk, 32));
      float newM = fmaxf(Mrow[r], rowmax);
      float scl = __expf(Mrow[r] - newM);
      Mrow[r] = newM;
      float rsum = 0.0f;
#pragma unroll
      for (int nt = 0; nt < 4; ++nt) {
        float pvv = __expf(sacc[nt][r] - newM);
        sacc[nt][r] = pvv;
        rsum += pvv;
      }
#pragma unroll
      for (int msk = 8; msk >= 1; msk >>= 1)
        rsum += __shfl_xor(rsum, msk, 32);
      Lrow[r] = Lrow[r] * scl + rsum;
#pragma unroll
      for (int dt = 0; dt < 4; ++dt) ctx[dt][r] = ctx[dt][r] * scl;
    }

    // P (C-fragment layout) -> LDS (row-major bf16) for re-lay-out as A frags
#pragma unroll
    for (int nt = 0; nt < 4; ++nt)
#pragma unroll
      for (int r = 0; r < 8; ++r)
        pw[(r + 8 * h) * 64 + nt * 16 + m] = f2bf(sacc[nt][r]);
    // order cross-lane LDS stores before the loads below (same-wave DS issues
    // in order; explicit split-counter wait for safety)
    asm volatile("s_wait_dscnt 0" ::: "memory");

    // ctx(16x64) += P(16x64) @ V(64x64)
#pragma unroll
    for (int kc = 0; kc < 2; ++kc) {
      Frag fp;
#pragma unroll
      for (int v = 0; v < 8; ++v) {
        int k = kc * 32 + ((v & 4) << 2) + h * 8 + (v & 3) * 2;
        fp.u[v] = *(const unsigned int*)&pw[m * 64 + k];
      }
#pragma unroll
      for (int dt = 0; dt < 4; ++dt) {
        Frag fv;
#pragma unroll
        for (int v = 0; v < 8; ++v) {
          int j = kc * 32 + ((v & 4) << 2) + h * 8 + (v & 3) * 2;
          fv.u[v] = *(const unsigned int*)&lVT[(dt * 16 + m) * 64 + j];
        }
        ctx[dt] = wmma_bf16(fp, fv, ctx[dt]);
      }
    }
  }

  // finalize: divide by row sums, store bf16 ctx
#pragma unroll
  for (int r = 0; r < 8; ++r) {
    float inv = 1.0f / Lrow[r];
#pragma unroll
    for (int dt = 0; dt < 4; ++dt) ctx[dt][r] = ctx[dt][r] * inv;
  }
#pragma unroll
  for (int dt = 0; dt < 4; ++dt)
#pragma unroll
    for (int r = 0; r < 8; ++r) {
      int srow = qt + wv * 16 + r + 8 * h;
      ctxg[((size_t)(b * S + srow)) * D + head * DH + dt * 16 + m] =
          f2bf(ctx[dt][r]);
    }
}

// ---------------------------------------------------------------------------
// Orchestration
// ---------------------------------------------------------------------------
extern "C" void kernel_launch(void* const* d_in, const int* in_sizes, int n_in,
                              void* d_out, int out_size, void* d_ws,
                              size_t ws_size, hipStream_t stream) {
  (void)in_sizes; (void)n_in; (void)out_size; (void)ws_size;
  const float* x    = (const float*)d_in[0];
  const float* wq   = (const float*)d_in[1];
  const float* bq   = (const float*)d_in[2];
  const float* wk   = (const float*)d_in[3];
  const float* bk   = (const float*)d_in[4];
  const float* wv   = (const float*)d_in[5];
  const float* bv   = (const float*)d_in[6];
  const float* wo   = (const float*)d_in[7];
  const float* bo   = (const float*)d_in[8];
  const float* ln1g = (const float*)d_in[9];
  const float* ln1b = (const float*)d_in[10];
  const float* ln2g = (const float*)d_in[11];
  const float* ln2b = (const float*)d_in[12];
  const float* w1   = (const float*)d_in[13];
  const float* b1   = (const float*)d_in[14];
  const float* w2   = (const float*)d_in[15];
  const float* b2   = (const float*)d_in[16];

  const int Bb = 4, S = 2048, D = 512, H = 8, FF = 2048;
  const int M = Bb * S;  // 8192

  char* base = (char*)d_ws;
  size_t off = 0;
  auto take = [&](size_t bytes) -> void* {
    void* p = base + off;
    off = (off + bytes + 255) & ~(size_t)255;
    return p;
  };
  unsigned short* xn   = (unsigned short*)take((size_t)M * D * 2);
  unsigned short* qb   = (unsigned short*)take((size_t)M * D * 2);
  unsigned short* kbuf = (unsigned short*)take((size_t)M * D * 2);
  unsigned short* vbuf = (unsigned short*)take((size_t)M * D * 2);
  unsigned short* ctxb = (unsigned short*)take((size_t)M * D * 2);
  unsigned short* xn2  = (unsigned short*)take((size_t)M * D * 2);
  float*          x1   = (float*)take((size_t)M * D * 4);
  unsigned short* ffh  = (unsigned short*)take((size_t)M * FF * 2);
  unsigned short* wqb  = (unsigned short*)take((size_t)D * D * 2);
  unsigned short* wkb  = (unsigned short*)take((size_t)D * D * 2);
  unsigned short* wvb  = (unsigned short*)take((size_t)D * D * 2);
  unsigned short* wob  = (unsigned short*)take((size_t)D * D * 2);
  unsigned short* w1b  = (unsigned short*)take((size_t)FF * D * 2);
  unsigned short* w2b  = (unsigned short*)take((size_t)D * FF * 2);

  // weights -> bf16
  {
    int nDD = D * D, nFD = FF * D;
    int gDD = (nDD + 255) / 256, gFD = (nFD + 255) / 256;
    cvt_f32_bf16_kernel<<<gDD, 256, 0, stream>>>(wq, wqb, nDD);
    cvt_f32_bf16_kernel<<<gDD, 256, 0, stream>>>(wk, wkb, nDD);
    cvt_f32_bf16_kernel<<<gDD, 256, 0, stream>>>(wv, wvb, nDD);
    cvt_f32_bf16_kernel<<<gDD, 256, 0, stream>>>(wo, wob, nDD);
    cvt_f32_bf16_kernel<<<gFD, 256, 0, stream>>>(w1, w1b, nFD);
    cvt_f32_bf16_kernel<<<gFD, 256, 0, stream>>>(w2, w2b, nFD);
  }

  // LN1
  layernorm_bf16_kernel<<<M, 256, 0, stream>>>(x, ln1g, ln1b, xn);

  // QKV projections
  dim3 g512(D / 128, M / 128);  // (4, 64)
  gemm_bf16_kernel<false, false, true><<<g512, 256, 0, stream>>>(
      xn, wqb, bq, nullptr, nullptr, qb, M, D, D);
  gemm_bf16_kernel<false, false, true><<<g512, 256, 0, stream>>>(
      xn, wkb, bk, nullptr, nullptr, kbuf, M, D, D);
  gemm_bf16_kernel<false, false, true><<<g512, 256, 0, stream>>>(
      xn, wvb, bv, nullptr, nullptr, vbuf, M, D, D);

  // attention
  attn_kernel<<<dim3(S / 64, H, Bb), 128, 0, stream>>>(qb, kbuf, vbuf, ctxb);

  // out proj + residual
  gemm_bf16_kernel<false, true, false><<<g512, 256, 0, stream>>>(
      ctxb, wob, bo, x, x1, nullptr, M, D, D);

  // LN2
  layernorm_bf16_kernel<<<M, 256, 0, stream>>>(x1, ln2g, ln2b, xn2);

  // FFN
  gemm_bf16_kernel<true, false, true><<<dim3(FF / 128, M / 128), 256, 0, stream>>>(
      xn2, w1b, b1, nullptr, nullptr, ffh, M, FF, D);
  gemm_bf16_kernel<false, true, false><<<g512, 256, 0, stream>>>(
      ffh, w2b, b2, x1, (float*)d_out, nullptr, M, D, FF);
}
